// MHAttention_30949534335122
// MI455X (gfx1250) — matrix operand order repaired and checked by reading, not compiled, not run on hardware
//
#include <hip/hip_runtime.h>

// ---------------------------------------------------------------------------
// Linformer multi-head attention for MI455X (gfx1250, wave32).
//   - All GEMMs via v_wmma_f32_16x16x32_bf16 (fp32 demoted to bf16 once,
//     f32 accumulation) — the matrix-core path CDNA5 rewards for fp32 data.
//   - 2x-unrolled ping-pong k-loops: two buffer sets refilled by loads
//     directly (no v_mov hand-off), WMMAs overlap in-flight loads.
//   - Fused scores->softmax->P*VF per 16-row tile; P never touches HBM.
//   - KE/VF operands staged into LDS with global_load_async_to_lds_b128
//     (ASYNCcnt) and shared by 4 row-tiles per block.
// ---------------------------------------------------------------------------

typedef __bf16 bf16;
typedef __attribute__((ext_vector_type(16))) __bf16 v16bf;
typedef __attribute__((ext_vector_type(8)))  float  v8f;

namespace {
constexpr int kB = 2;      // batch
constexpr int kN = 4096;   // sequence length
constexpr int kC = 1024;   // channels
constexpr int kH = 16;     // heads
constexpr int kD = 64;     // head dim
constexpr int kK = 256;    // low-rank dim
constexpr int kF = kH * kD; // 1024 (concat feature dim)
}

// ---- WMMA fragment index maps (per CDNA5 ISA 7.12.2, wave32) ---------------
__device__ __forceinline__ int frag_a_k(int lane, int e) {
  return ((((e >> 1) & 3) << 1) | ((e >> 3) << 4) | (((lane >> 4) & 1) << 3) | (e & 1));
}
__device__ __forceinline__ int frag_b_k(int lane, int e) {
  return ((e & 7) | (((lane >> 4) & 1) << 3) | ((e >> 3) << 4));
}
__device__ __forceinline__ int frag_c_m(int lane, int r) {
  return r + (((lane >> 4) & 1) << 3);
}

__device__ __forceinline__ v8f wmma_bf16(v16bf a, v16bf b, v8f c) {
  return __builtin_amdgcn_wmma_f32_16x16x32_bf16(false, a, false, b, (short)0, c,
                                                 false, false);
}

// Async bulk copy global -> LDS, 16B per lane per issue (ASYNCcnt tracked).
__device__ __forceinline__ void async_copy_b128(unsigned lds_addr,
                                                unsigned long long gaddr) {
  asm volatile("global_load_async_to_lds_b128 %0, %1, off"
               :: "v"(lds_addr), "v"(gaddr) : "memory");
}
__device__ __forceinline__ void wait_asynccnt0() {
  asm volatile("s_wait_asynccnt 0" ::: "memory");
}
__device__ __forceinline__ unsigned lds_lo32(const void* p) {
  // LDS aperture: generic addr[31:0] is the LDS byte offset.
  return (unsigned)(unsigned long long)p;
}

// ---- fp32 -> bf16 conversion ------------------------------------------------
__global__ void cvt_f32_bf16(const float* __restrict__ in, bf16* __restrict__ out,
                             size_t n) {
  size_t i = (size_t)blockIdx.x * blockDim.x + threadIdx.x;
  size_t stride = (size_t)gridDim.x * blockDim.x;
  for (; i < n; i += stride) out[i] = (bf16)in[i];
}

// Consume one fragment-set: 4 WMMAs sharing an A fragment.
#define WMMA_CONSUME(AF, BF)                                        \
  do {                                                              \
    _Pragma("unroll") for (int t = 0; t < 4; ++t)                   \
        acc[t] = wmma_bf16(AF, BF[t], acc[t]);                      \
  } while (0)

// ---- QKV projection: Out[b,h,n,d] = sum_c X[b,n,c] * W[h,d,c] --------------
__global__ void __launch_bounds__(128)
qkv_proj(const bf16* __restrict__ X,   // [B,N,C]
         const bf16* __restrict__ W,   // [H,D,C]
         bf16* __restrict__ Out) {     // [B,H,N,D]
  const int lane = threadIdx.x & 31;
  const int wave = threadIdx.x >> 5;
  const int rt   = blockIdx.x * 4 + wave;   // 16-row tile (0..255)
  const int h    = blockIdx.y;
  const int b    = blockIdx.z;
  const int row0 = rt * 16;
  const int am = lane & 15, bn = lane & 15;

  const bf16* Xb = X + ((size_t)b * kN + row0) * kC;
  const bf16* Wh = W + (size_t)h * kD * kC;

  auto loadA = [&](int c0) {
    v16bf a;
#pragma unroll
    for (int e = 0; e < 16; ++e)
      a[e] = Xb[(size_t)am * kC + c0 + frag_a_k(lane, e)];
    return a;
  };
  auto loadB = [&](int c0, int t) {
    v16bf v;
#pragma unroll
    for (int e = 0; e < 16; ++e)
      v[e] = Wh[(size_t)(t * 16 + bn) * kC + c0 + frag_b_k(lane, e)];
    return v;
  };

  v8f acc[4] = {};
  // ping-pong buffers, refilled by loads directly (no register hand-off)
  v16bf pA = loadA(0);
  v16bf pB[4] = {loadB(0, 0), loadB(0, 1), loadB(0, 2), loadB(0, 3)};
  v16bf qA = loadA(32);
  v16bf qB[4] = {loadB(32, 0), loadB(32, 1), loadB(32, 2), loadB(32, 3)};
  for (int c0 = 64; c0 < kC; c0 += 64) {
    WMMA_CONSUME(pA, pB);
    pA = loadA(c0);
#pragma unroll
    for (int t = 0; t < 4; ++t) pB[t] = loadB(c0, t);
    WMMA_CONSUME(qA, qB);
    qA = loadA(c0 + 32);
#pragma unroll
    for (int t = 0; t < 4; ++t) qB[t] = loadB(c0 + 32, t);
  }
  WMMA_CONSUME(pA, pB);
  WMMA_CONSUME(qA, qB);

  bf16* Ob = Out + (((size_t)b * kH + h) * kN + row0) * kD;
#pragma unroll
  for (int t = 0; t < 4; ++t)
#pragma unroll
    for (int r = 0; r < 8; ++r)
      Ob[(size_t)frag_c_m(lane, r) * kD + t * 16 + bn] = (bf16)acc[t][r];
}

// ---- KE[b,h,d,k] = sum_n Kh[b,h,n,d] * We[k,n] + be[k] ---------------------
__global__ void __launch_bounds__(128)
ke_proj(const bf16* __restrict__ Kh,   // [B,H,N,D]
        const bf16* __restrict__ We,   // [K,N]
        const float* __restrict__ be,  // [K]
        bf16* __restrict__ KE) {       // [B,H,D,K]
  const int lane = threadIdx.x & 31;
  const int wave = threadIdx.x >> 5;
  const int h = blockIdx.y, b = blockIdx.z;
  const int d0 = wave * 16;
  const int kg = blockIdx.x;            // 64 k_lr columns per block
  const int am = lane & 15, bn = lane & 15;

  const bf16* Kb = Kh + ((size_t)b * kH + h) * kN * kD;

  auto loadA = [&](int c0) {            // A[d, n] = Kh[n, d]
    v16bf a;
#pragma unroll
    for (int e = 0; e < 16; ++e)
      a[e] = Kb[(size_t)(c0 + frag_a_k(lane, e)) * kD + d0 + am];
    return a;
  };
  auto loadB = [&](int c0, int t) {     // B[n, k_lr] = We[k_lr, n]
    const int kl0 = kg * 64 + t * 16;
    v16bf v;
#pragma unroll
    for (int e = 0; e < 16; ++e)
      v[e] = We[(size_t)(kl0 + bn) * kN + c0 + frag_b_k(lane, e)];
    return v;
  };

  v8f acc[4] = {};
  v16bf pA = loadA(0);
  v16bf pB[4] = {loadB(0, 0), loadB(0, 1), loadB(0, 2), loadB(0, 3)};
  v16bf qA = loadA(32);
  v16bf qB[4] = {loadB(32, 0), loadB(32, 1), loadB(32, 2), loadB(32, 3)};
  for (int c0 = 64; c0 < kN; c0 += 64) {
    WMMA_CONSUME(pA, pB);
    pA = loadA(c0);
#pragma unroll
    for (int t = 0; t < 4; ++t) pB[t] = loadB(c0, t);
    WMMA_CONSUME(qA, qB);
    qA = loadA(c0 + 32);
#pragma unroll
    for (int t = 0; t < 4; ++t) qB[t] = loadB(c0 + 32, t);
  }
  WMMA_CONSUME(pA, pB);
  WMMA_CONSUME(qA, qB);

  bf16* Ob = KE + ((size_t)b * kH + h) * kD * kK;
#pragma unroll
  for (int t = 0; t < 4; ++t)
#pragma unroll
    for (int r = 0; r < 8; ++r) {
      const int kcol = kg * 64 + t * 16 + bn;
      Ob[(size_t)(d0 + frag_c_m(lane, r)) * kK + kcol] =
          (bf16)(acc[t][r] + be[kcol]);
    }
}

// ---- VF[b,h,k,d] = sum_n V[b,h,n,d] * Wf[k,n] + bf[k] ----------------------
__global__ void __launch_bounds__(128)
vf_proj(const bf16* __restrict__ V,    // [B,H,N,D]
        const bf16* __restrict__ Wf,   // [K,N]
        const float* __restrict__ bfv, // [K]
        bf16* __restrict__ VF) {       // [B,H,K,D]
  const int lane = threadIdx.x & 31;
  const int wave = threadIdx.x >> 5;
  const int h = blockIdx.y, b = blockIdx.z;
  const int m0 = (blockIdx.x * 4 + wave) * 16;  // k_lr row tile
  const int am = lane & 15, bn = lane & 15;

  const bf16* Vb = V + ((size_t)b * kH + h) * kN * kD;

  auto loadA = [&](int c0) {            // A[k_lr, n] = Wf[k_lr, n]
    v16bf a;
#pragma unroll
    for (int e = 0; e < 16; ++e)
      a[e] = Wf[(size_t)(m0 + am) * kN + c0 + frag_a_k(lane, e)];
    return a;
  };
  auto loadB = [&](int c0, int t) {     // B[n, d] = V[n, d]
    v16bf v;
#pragma unroll
    for (int e = 0; e < 16; ++e)
      v[e] = Vb[(size_t)(c0 + frag_b_k(lane, e)) * kD + t * 16 + bn];
    return v;
  };

  v8f acc[4] = {};
  v16bf pA = loadA(0);
  v16bf pB[4] = {loadB(0, 0), loadB(0, 1), loadB(0, 2), loadB(0, 3)};
  v16bf qA = loadA(32);
  v16bf qB[4] = {loadB(32, 0), loadB(32, 1), loadB(32, 2), loadB(32, 3)};
  for (int c0 = 64; c0 < kN; c0 += 64) {
    WMMA_CONSUME(pA, pB);
    pA = loadA(c0);
#pragma unroll
    for (int t = 0; t < 4; ++t) pB[t] = loadB(c0, t);
    WMMA_CONSUME(qA, qB);
    qA = loadA(c0 + 32);
#pragma unroll
    for (int t = 0; t < 4; ++t) qB[t] = loadB(c0 + 32, t);
  }
  WMMA_CONSUME(pA, pB);
  WMMA_CONSUME(qA, qB);

  bf16* Ob = VF + ((size_t)b * kH + h) * kK * kD;
#pragma unroll
  for (int t = 0; t < 4; ++t)
#pragma unroll
    for (int r = 0; r < 8; ++r) {
      const int m = m0 + frag_c_m(lane, r);
      Ob[(size_t)m * kD + t * 16 + bn] = (bf16)(acc[t][r] + bfv[m]);
    }
}

// ---- Fused attention ------------------------------------------------------
// grid (N/16/4, H, B), block 128: 4 waves, one 16-row tile each, sharing
// async-LDS-staged KE[64x256] and VF[256x64] for this (b,h).
__global__ void __launch_bounds__(128)
linattn(const bf16* __restrict__ Q,    // [B,H,N,D]
        const bf16* __restrict__ KE,   // [B,H,D,K]
        const bf16* __restrict__ VF,   // [B,H,K,D]
        bf16* __restrict__ Concat) {   // [B,N,H*D]
  __shared__ bf16  sKE[kD][kK];       // 32 KB
  __shared__ bf16  sVF[kK][kD];       // 32 KB
  __shared__ float sS[4][16][kK];     // 64 KB (scores; reused in place for P)

  const int lane = threadIdx.x & 31;
  const int wave = threadIdx.x >> 5;
  const int rt = blockIdx.x * 4 + wave, h = blockIdx.y, b = blockIdx.z;
  const int row0 = rt * 16;
  const int am = lane & 15, bn = lane & 15;

  const bf16* Qb  = Q  + (((size_t)b * kH + h) * kN + row0) * kD;
  const bf16* KEb = KE + ((size_t)b * kH + h) * kD * kK;
  const bf16* VFb = VF + ((size_t)b * kH + h) * kK * kD;

  // ---- async stage KE and VF into LDS (2048 x 16B each, 128 lanes) ----
  {
    const unsigned lke = lds_lo32(&sKE[0][0]);
    const unsigned lvf = lds_lo32(&sVF[0][0]);
    const unsigned long long gke = (unsigned long long)KEb;
    const unsigned long long gvf = (unsigned long long)VFb;
#pragma unroll
    for (int i = 0; i < 16; ++i) {
      const unsigned off = (threadIdx.x + i * 128) * 16;
      async_copy_b128(lke + off, gke + off);
      async_copy_b128(lvf + off, gvf + off);
    }
    wait_asynccnt0();
  }
  __syncthreads();

  float (&S)[16][kK] = sS[wave];

  // scores[16 x 256] = Q_tile[16x64] * KE[64x256] * (1/sqrt(D))
  for (int g = 0; g < 4; ++g) {
    v8f acc[4] = {};
#pragma unroll
    for (int d0 = 0; d0 < kD; d0 += 32) {
      v16bf a;
#pragma unroll
      for (int e = 0; e < 16; ++e)
        a[e] = Qb[(size_t)am * kD + d0 + frag_a_k(lane, e)];
#pragma unroll
      for (int t = 0; t < 4; ++t) {
        const int kl0 = g * 64 + t * 16;
        v16bf bm;
#pragma unroll
        for (int e = 0; e < 16; ++e)
          bm[e] = sKE[d0 + frag_b_k(lane, e)][kl0 + bn];
        acc[t] = wmma_bf16(a, bm, acc[t]);
      }
    }
#pragma unroll
    for (int t = 0; t < 4; ++t)
#pragma unroll
      for (int r = 0; r < 8; ++r)
        S[frag_c_m(lane, r)][g * 64 + t * 16 + bn] = acc[t][r] * 0.125f;
  }
  __syncthreads();

  // per-row softmax (lanes 0..15 of each wave); P written in place as bf16
  // (bf16 write at byte 2j never overtakes the f32 read at byte 4j).
  if (lane < 16) {
    float mx = -3.4e38f;
    for (int j = 0; j < kK; ++j) mx = fmaxf(mx, S[lane][j]);
    float sum = 0.f;
    for (int j = 0; j < kK; ++j) {
      float e = __expf(S[lane][j] - mx);
      S[lane][j] = e;
      sum += e;
    }
    const float inv = 1.f / sum;
    bf16* pr = reinterpret_cast<bf16*>(&S[lane][0]);
    for (int j = 0; j < kK; ++j) pr[j] = (bf16)(S[lane][j] * inv);
  }
  __syncthreads();

  // out_tile[16x64] = P[16x256] * VF[256x64]
  bf16* Ob = Concat + ((size_t)b * kN + row0) * kF + (size_t)h * kD;
#pragma unroll
  for (int t = 0; t < 4; ++t) {
    v8f acc = {};
#pragma unroll
    for (int k0 = 0; k0 < kK; k0 += 32) {
      v16bf a;
      const bf16* pr = reinterpret_cast<const bf16*>(&S[am][0]);
#pragma unroll
      for (int e = 0; e < 16; ++e)
        a[e] = pr[k0 + frag_a_k(lane, e)];
      v16bf bm;
#pragma unroll
      for (int e = 0; e < 16; ++e)
        bm[e] = sVF[k0 + frag_b_k(lane, e)][t * 16 + bn];
      acc = wmma_bf16(a, bm, acc);
    }
#pragma unroll
    for (int r = 0; r < 8; ++r)
      Ob[(size_t)frag_c_m(lane, r) * kF + t * 16 + bn] = (bf16)acc[r];
  }
}

// ---- Output projection: out[b,n,c] = sum_f Concat[b,n,f]*Wo[c,f] + bo[c] ---
__global__ void __launch_bounds__(128)
out_proj(const bf16* __restrict__ A,   // [B,N,F]
         const bf16* __restrict__ Wo,  // [C,F]
         const float* __restrict__ bo, // [C]
         float* __restrict__ out) {    // [B,N,C]
  const int lane = threadIdx.x & 31;
  const int wave = threadIdx.x >> 5;
  const int row0 = (blockIdx.y * 4 + wave) * 16;
  const int cg = blockIdx.x;            // 64 output cols per block
  const int b = blockIdx.z;
  const int am = lane & 15, bn = lane & 15;

  const bf16* Ab = A + ((size_t)b * kN + row0) * kF;

  auto loadA = [&](int f0) {
    v16bf a;
#pragma unroll
    for (int e = 0; e < 16; ++e)
      a[e] = Ab[(size_t)am * kF + f0 + frag_a_k(lane, e)];
    return a;
  };
  auto loadB = [&](int f0, int t) {     // B[f, c] = Wo[c, f]
    const int c0 = cg * 64 + t * 16;
    v16bf v;
#pragma unroll
    for (int e = 0; e < 16; ++e)
      v[e] = Wo[(size_t)(c0 + bn) * kF + f0 + frag_b_k(lane, e)];
    return v;
  };

  v8f acc[4] = {};
  v16bf pA = loadA(0);
  v16bf pB[4] = {loadB(0, 0), loadB(0, 1), loadB(0, 2), loadB(0, 3)};
  v16bf qA = loadA(32);
  v16bf qB[4] = {loadB(32, 0), loadB(32, 1), loadB(32, 2), loadB(32, 3)};
  for (int f0 = 64; f0 < kF; f0 += 64) {
    WMMA_CONSUME(pA, pB);
    pA = loadA(f0);
#pragma unroll
    for (int t = 0; t < 4; ++t) pB[t] = loadB(f0, t);
    WMMA_CONSUME(qA, qB);
    qA = loadA(f0 + 32);
#pragma unroll
    for (int t = 0; t < 4; ++t) qB[t] = loadB(f0 + 32, t);
  }
  WMMA_CONSUME(pA, pB);
  WMMA_CONSUME(qA, qB);

#pragma unroll
  for (int t = 0; t < 4; ++t)
#pragma unroll
    for (int r = 0; r < 8; ++r) {
      const int c = cg * 64 + t * 16 + bn;
      out[((size_t)b * kN + row0 + frag_c_m(lane, r)) * kC + c] =
          acc[t][r] + bo[c];
    }
}

// ---------------------------------------------------------------------------
extern "C" void kernel_launch(void* const* d_in, const int* in_sizes, int n_in,
                              void* d_out, int out_size, void* d_ws, size_t ws_size,
                              hipStream_t stream) {
  const float* tensor = (const float*)d_in[0];
  const float* Wq = (const float*)d_in[1];
  const float* Wk = (const float*)d_in[2];
  const float* Wv = (const float*)d_in[3];
  const float* We = (const float*)d_in[4];
  const float* be = (const float*)d_in[5];
  const float* Wf = (const float*)d_in[6];
  const float* bfv = (const float*)d_in[7];
  const float* Wo = (const float*)d_in[8];
  const float* bo = (const float*)d_in[9];
  float* out = (float*)d_out;

  // Workspace layout (bf16, 256B-aligned slabs), ~95 MB total.
  bf16* base = (bf16*)d_ws;
  size_t o = 0;
  auto alloc = [&](size_t n) {
    bf16* p = base + o;
    o += (n + 127) & ~(size_t)127;
    return p;
  };
  bf16* Xb  = alloc((size_t)kB * kN * kC);
  bf16* Wqb = alloc((size_t)kH * kD * kC);
  bf16* Wkb = alloc((size_t)kH * kD * kC);
  bf16* Wvb = alloc((size_t)kH * kD * kC);
  bf16* Web = alloc((size_t)kK * kN);
  bf16* Wfb = alloc((size_t)kK * kN);
  bf16* Wob = alloc((size_t)kC * kF);
  bf16* Qb  = alloc((size_t)kB * kH * kN * kD);
  bf16* Kb  = alloc((size_t)kB * kH * kN * kD);
  bf16* Vb  = alloc((size_t)kB * kH * kN * kD);
  bf16* KEb = alloc((size_t)kB * kH * kD * kK);
  bf16* VFb = alloc((size_t)kB * kH * kK * kD);
  bf16* Cb  = alloc((size_t)kB * kN * kF);
  if (o * sizeof(bf16) > ws_size) return;  // insufficient scratch

  auto cvt = [&](const float* s, bf16* d, size_t n) {
    cvt_f32_bf16<<<dim3((unsigned)((n + 255) / 256)), dim3(256), 0, stream>>>(s, d, n);
  };
  cvt(tensor, Xb, (size_t)kB * kN * kC);
  cvt(Wq, Wqb, (size_t)kH * kD * kC);
  cvt(Wk, Wkb, (size_t)kH * kD * kC);
  cvt(Wv, Wvb, (size_t)kH * kD * kC);
  cvt(We, Web, (size_t)kK * kN);
  cvt(Wf, Wfb, (size_t)kK * kN);
  cvt(Wo, Wob, (size_t)kC * kF);

  const dim3 gq(kN / 16 / 4, kH, kB);          // 64,16,2
  qkv_proj<<<gq, 128, 0, stream>>>(Xb, Wqb, Qb);
  qkv_proj<<<gq, 128, 0, stream>>>(Xb, Wkb, Kb);
  qkv_proj<<<gq, 128, 0, stream>>>(Xb, Wvb, Vb);

  ke_proj<<<dim3(4, kH, kB), 128, 0, stream>>>(Kb, Web, be, KEb);
  vf_proj<<<dim3(4, kH, kB), 128, 0, stream>>>(Vb, Wfb, bfv, VFb);

  linattn<<<dim3(kN / 16 / 4, kH, kB), 128, 0, stream>>>(Qb, KEb, VFb, Cb);

  out_proj<<<dim3(kC / 64, kN / 16 / 4, kB), 128, 0, stream>>>(Cb, Wob, bo, out);
}
